// MoeLayer_24223615549788
// MI455X (gfx1250) — matrix operand order repaired and checked by reading, not compiled, not run on hardware
//
#include <hip/hip_runtime.h>

// ---------------- problem constants (from reference) ----------------
namespace {
constexpr int G_GROUPS = 2;
constexpr int T_TOK    = 4096;              // tokens per group
constexpr int GT       = G_GROUPS * T_TOK;  // 8192 tokens total
constexpr int H_DIM    = 1024;
constexpr int E_EXP    = 8;
constexpr int F_DIM    = 4096;
constexpr int CAP      = 640;               // expert capacity
constexpr int SLOTS    = E_EXP * G_GROUPS * CAP; // 10240 expert rows

// GEMM tiling: block 128x256, 8 waves in 2(M)x4(N), each wave 64x64 = 4x4 WMMA frags
constexpr int BM = 128, BN = 256, BK = 32;
constexpr int LDSS = 40;                    // padded LDS row stride in bf16 elems (80B, 16B-aligned)
}

typedef __attribute__((ext_vector_type(16))) __bf16 bf16x16;
typedef __attribute__((ext_vector_type(2)))  __bf16 bf16x2;
typedef __attribute__((ext_vector_type(8)))  float  f32x8;

// native converts -> v_cvt_pk_bf16_f32 / v_cvt_bf16_f32 (RNE in hardware)
static __device__ __forceinline__ unsigned short f2bf(float f) {
    return __builtin_bit_cast(unsigned short, (__bf16)f);
}
static __device__ __forceinline__ unsigned pack_bf16(float lo, float hi) {
    bf16x2 v = { (__bf16)lo, (__bf16)hi };
    return __builtin_bit_cast(unsigned, v);
}

// ---------------- 1) init slot tables ----------------
__global__ void init_kernel(int* __restrict__ slot_token, int* __restrict__ token_assign) {
    int i = blockIdx.x * blockDim.x + threadIdx.x;
    if (i < SLOTS)  slot_token[i]   = -1;
    if (i < GT * 2) token_assign[i] = -1;
}

// ---------------- 2) router: logits -> softmax -> top2 ----------------
__global__ void router_kernel(const float* __restrict__ x, const float* __restrict__ wr,
                              int* __restrict__ topk, float2* __restrict__ gates) {
    const int lane = threadIdx.x & 31, wid = threadIdx.x >> 5;
    const int t = blockIdx.x * 8 + wid;
    if (t >= GT) return;
    float p[E_EXP];
    #pragma unroll
    for (int e = 0; e < E_EXP; ++e) p[e] = 0.f;
    const float* xr = x + (size_t)t * H_DIM;
    for (int h = lane; h < H_DIM; h += 32) {
        float xv = xr[h];
        const float4* w = reinterpret_cast<const float4*>(wr + (size_t)h * E_EXP);
        float4 w0 = w[0], w1 = w[1];
        p[0] += xv * w0.x; p[1] += xv * w0.y; p[2] += xv * w0.z; p[3] += xv * w0.w;
        p[4] += xv * w1.x; p[5] += xv * w1.y; p[6] += xv * w1.z; p[7] += xv * w1.w;
    }
    #pragma unroll
    for (int e = 0; e < E_EXP; ++e)
        for (int off = 16; off > 0; off >>= 1)
            p[e] += __shfl_xor(p[e], off, 32);
    if (lane == 0) {
        float m = p[0];
        #pragma unroll
        for (int e = 1; e < E_EXP; ++e) m = fmaxf(m, p[e]);
        float s = 0.f, q[E_EXP];
        #pragma unroll
        for (int e = 0; e < E_EXP; ++e) { q[e] = __expf(p[e] - m); s += q[e]; }
        float inv = 1.f / s;
        #pragma unroll
        for (int e = 0; e < E_EXP; ++e) q[e] *= inv;
        int e0 = 0;
        #pragma unroll
        for (int e = 1; e < E_EXP; ++e) if (q[e] > q[e0]) e0 = e;
        int e1 = (e0 == 0) ? 1 : 0;
        #pragma unroll
        for (int e = 0; e < E_EXP; ++e) if (e != e0 && q[e] > q[e1]) e1 = e;
        topk[t]  = e0 | (e1 << 8);
        gates[t] = make_float2(q[e0], q[e1]);
    }
}

// ---------------- 3) capacity assignment (deterministic priority scan) ----------------
__global__ void assign_kernel(const int* __restrict__ topk,
                              int* __restrict__ slot_token, int* __restrict__ token_assign) {
    const int e = blockIdx.x & (E_EXP - 1);
    const int g = blockIdx.x >> 3;
    const int lane = threadIdx.x;
    int base = 0;
    for (int k = 0; k < 2; ++k) {
        for (int t0 = 0; t0 < T_TOK; t0 += 32) {
            const int t = t0 + lane;
            const int pack = topk[g * T_TOK + t];
            const int ek = (pack >> (k * 8)) & 0xFF;
            const bool match = (ek == e);
            const unsigned bal = (unsigned)__ballot(match);
            const int prefix = __popc(bal & ((1u << lane) - 1u));
            if (match) {
                const int pos = base + prefix;
                if (pos < CAP) {
                    const int s = (e * G_GROUPS + g) * CAP + pos;   // expert row index
                    slot_token[s] = g * T_TOK + t;                  // global token row
                    token_assign[(g * T_TOK + t) * 2 + k] = s;
                }
            }
            base += __popc(bal);
        }
    }
}

// ---------------- 4) gather tokens into bf16 expert buffers ----------------
__global__ void gather_kernel(const float* __restrict__ x, const int* __restrict__ slot_token,
                              unsigned short* __restrict__ expert_in) {
    const int s = blockIdx.x;
    const int t = slot_token[s];
    unsigned short* dst = expert_in + (size_t)s * H_DIM;
    if (t >= 0) {
        const float* src = x + (size_t)t * H_DIM;
        for (int h = threadIdx.x; h < H_DIM; h += blockDim.x) dst[h] = f2bf(src[h]);
    } else {
        for (int h = threadIdx.x; h < H_DIM; h += blockDim.x) dst[h] = 0;
    }
}

// ---------------- 5/6) batched WMMA GEMM, double-buffered LDS ----------------
// A: bf16 [M,K] (lda=K), B: fp32 [K,N] (ldb=N) converted+transposed to bf16 [n][k] in LDS.
// Block tile 128x256, 8 waves (2Mx4N), wave tile 64x64 = 16 v_wmma_f32_16x16x32_bf16 per k-step.
template <bool RELU, bool OUT_BF16>
__global__ __launch_bounds__(256)
void wmma_gemm_kernel(const unsigned short* __restrict__ A, const float* __restrict__ B,
                      void* __restrict__ Cout, int M, int N, int K,
                      size_t strideA, size_t strideB, size_t strideC) {
    __shared__ unsigned short As[2][BM * LDSS];   // [m][k] bf16, double buffered
    __shared__ unsigned short Bs[2][BN * LDSS];   // transposed [n][k] bf16, double buffered
    (void)M;

    const int tid  = threadIdx.x;
    const int lane = tid & 31, wid = tid >> 5;
    const int waveM = wid & 1, waveN = wid >> 1;     // 2x4 wave grid, 64x64 per wave
    const int eg = blockIdx.z;                       // e*G + g
    const int mBlock = blockIdx.y * BM;
    const int nBlock = blockIdx.x * BN;

    const unsigned short* Ag = A + strideA * eg + (size_t)mBlock * K;
    const float*          Bg = B + strideB * (size_t)(eg >> 1) + nBlock;   // e = eg/2 (G==2)

    // loader mapping: A = 128 rows x two 16-elem halves; B = 16 k-pairs x 16 col-segments of 16
    const int rowA = tid >> 1, halfA = tid & 1;
    const int kpB  = tid >> 4, nsegB = tid & 15;
    const unsigned short* aSrc = Ag + (size_t)rowA * K + halfA * 16;
    const float*          bSrc = Bg + (size_t)(2 * kpB) * N + nsegB * 16;

    f32x8 acc[4][4];
    #pragma unroll
    for (int i = 0; i < 4; ++i)
        #pragma unroll
        for (int j = 0; j < 4; ++j)
            #pragma unroll
            for (int r = 0; r < 8; ++r) acc[i][j][r] = 0.f;

    const int mr = lane & 15;
    const int kA = (lane >= 16) ? 8 : 0;    // A: K halves per ISA 16-bit A layout
    const int kB = (lane >= 16) ? 8 : 0;    // B: K halves per ISA 16-bit B layout

    // ---- helpers ----
    auto storeTile = [&](int buf, uint4 ta0, uint4 ta1,
                         const float4* fr0, const float4* fr1) {
        uint4* d = reinterpret_cast<uint4*>(&As[buf][rowA * LDSS + halfA * 16]);
        d[0] = ta0; d[1] = ta1;
        #pragma unroll
        for (int q = 0; q < 4; ++q) {
            const float r0[4] = { fr0[q].x, fr0[q].y, fr0[q].z, fr0[q].w };
            const float r1[4] = { fr1[q].x, fr1[q].y, fr1[q].z, fr1[q].w };
            #pragma unroll
            for (int j = 0; j < 4; ++j)   // pack (k, k+1) -> one dword store into [n][k]
                *reinterpret_cast<unsigned*>(
                    &Bs[buf][(nsegB * 16 + q * 4 + j) * LDSS + 2 * kpB]) =
                    pack_bf16(r0[j], r1[j]);
        }
    };

    union Frag { uint4 u[2]; bf16x16 v; };
    auto computeStep = [&](int buf) {
        Frag bfb[4];
        #pragma unroll
        for (int ni = 0; ni < 4; ++ni) {
            const unsigned short* p = &Bs[buf][(waveN * 64 + ni * 16 + mr) * LDSS + kB];
            bfb[ni].u[0] = *reinterpret_cast<const uint4*>(p);            // K = kB..kB+7
            bfb[ni].u[1] = *reinterpret_cast<const uint4*>(p + 16);       // K = kB+16..kB+23
        }
        #pragma unroll
        for (int mi = 0; mi < 4; ++mi) {
            Frag af;
            const unsigned short* p = &As[buf][(waveM * 64 + mi * 16 + mr) * LDSS];
            af.u[0] = *reinterpret_cast<const uint4*>(p + kA);            // K = kA..kA+7
            af.u[1] = *reinterpret_cast<const uint4*>(p + kA + 16);       // K = kA+16..kA+23
            #pragma unroll
            for (int ni = 0; ni < 4; ++ni)
                acc[mi][ni] = __builtin_amdgcn_wmma_f32_16x16x32_bf16(
                    false, af.v, false, bfb[ni].v, (short)0, acc[mi][ni], false, false);
        }
    };

    // ---- prologue: tile 0 -> buffer 0 ----
    {
        uint4 a0 = *reinterpret_cast<const uint4*>(aSrc);
        uint4 a1 = *reinterpret_cast<const uint4*>(aSrc + 8);
        float4 r0[4], r1[4];
        #pragma unroll
        for (int q = 0; q < 4; ++q) {
            r0[q] = *reinterpret_cast<const float4*>(bSrc + q * 4);
            r1[q] = *reinterpret_cast<const float4*>(bSrc + N + q * 4);
        }
        storeTile(0, a0, a1, r0, r1);
    }
    __syncthreads();

    // ---- steady state: load next / compute current / store next ----
    const int nSteps = K / BK;
    int s = 0;
    for (; s + 1 < nSteps; ++s) {
        const unsigned short* ap = aSrc + (size_t)(s + 1) * BK;
        uint4 na0 = *reinterpret_cast<const uint4*>(ap);
        uint4 na1 = *reinterpret_cast<const uint4*>(ap + 8);
        const float* bp = bSrc + (size_t)(s + 1) * BK * N;
        float4 nr0[4], nr1[4];
        #pragma unroll
        for (int q = 0; q < 4; ++q) {
            nr0[q] = *reinterpret_cast<const float4*>(bp + q * 4);
            nr1[q] = *reinterpret_cast<const float4*>(bp + N + q * 4);
        }
        if (s + 2 < nSteps) {   // keep the fp32 weight stream ahead in L2
            __builtin_prefetch(bSrc + (size_t)(s + 2) * BK * N, 0, 0);
            __builtin_prefetch(aSrc + (size_t)(s + 2) * BK, 0, 0);
        }

        computeStep(s & 1);
        storeTile((s + 1) & 1, na0, na1, nr0, nr1);
        __syncthreads();
    }
    computeStep(s & 1);   // final tile, no staging

    // epilogue: C/D 16x16 f32 layout -> VGPR r holds M = r (+8 for lanes>=16), N = lane&15
    const int nOff = lane & 15;
    const int rOff = (lane >= 16) ? 8 : 0;
    #pragma unroll
    for (int mi = 0; mi < 4; ++mi)
        #pragma unroll
        for (int ni = 0; ni < 4; ++ni)
            #pragma unroll
            for (int r = 0; r < 8; ++r) {
                const int m = mBlock + waveM * 64 + mi * 16 + rOff + r;
                const int n = nBlock + waveN * 64 + ni * 16 + nOff;
                float v = acc[mi][ni][r];
                if (RELU) v = fmaxf(v, 0.f);
                const size_t idx = strideC * eg + (size_t)m * N + n;
                if (OUT_BF16) reinterpret_cast<unsigned short*>(Cout)[idx] = f2bf(v);
                else          reinterpret_cast<float*>(Cout)[idx] = v;
            }
}

// ---------------- 7) combine: y[t] = sum_k gate_k * expert_out[slot_k] ----------------
__global__ void combine_kernel(const int* __restrict__ token_assign, const float2* __restrict__ gates,
                               const float* __restrict__ expert_out, float* __restrict__ y) {
    const int gt = blockIdx.x;
    const int a0 = token_assign[gt * 2 + 0];
    const int a1 = token_assign[gt * 2 + 1];
    const float2 gv = gates[gt];
    float* dst = y + (size_t)gt * H_DIM;
    for (int h = threadIdx.x; h < H_DIM; h += blockDim.x) {
        float acc = 0.f;
        if (a0 >= 0) acc += gv.x * expert_out[(size_t)a0 * H_DIM + h];
        if (a1 >= 0) acc += gv.y * expert_out[(size_t)a1 * H_DIM + h];
        dst[h] = acc;
    }
}

// ---------------- host launcher ----------------
extern "C" void kernel_launch(void* const* d_in, const int* in_sizes, int n_in,
                              void* d_out, int out_size, void* d_ws, size_t ws_size,
                              hipStream_t stream) {
    (void)in_sizes; (void)n_in; (void)out_size; (void)ws_size;
    const float* x  = (const float*)d_in[0];   // [B,S,H] == [G*T, H]
    const float* wr = (const float*)d_in[1];   // [H, E]
    const float* wi = (const float*)d_in[2];   // [E, H, F]
    const float* wo = (const float*)d_in[3];   // [E, F, H]
    float* y = (float*)d_out;                  // [G*T, H]

    char* ws = (char*)d_ws;
    size_t off = 0;
    auto take = [&](size_t bytes) -> void* {
        void* p = ws + off;
        off = (off + bytes + 255) & ~(size_t)255;
        return p;
    };
    int*            topk         = (int*)take(sizeof(int) * GT);
    float2*         gates        = (float2*)take(sizeof(float2) * GT);
    int*            slot_token   = (int*)take(sizeof(int) * SLOTS);
    int*            token_assign = (int*)take(sizeof(int) * GT * 2);
    unsigned short* expert_in    = (unsigned short*)take(sizeof(unsigned short) * (size_t)SLOTS * H_DIM);
    unsigned short* hbuf         = (unsigned short*)take(sizeof(unsigned short) * (size_t)SLOTS * F_DIM);
    float*          expert_out   = (float*)take(sizeof(float) * (size_t)SLOTS * H_DIM);

    init_kernel<<<64, 256, 0, stream>>>(slot_token, token_assign);
    router_kernel<<<GT / 8, 256, 0, stream>>>(x, wr, topk, gates);
    assign_kernel<<<E_EXP * G_GROUPS, 32, 0, stream>>>(topk, slot_token, token_assign);
    gather_kernel<<<SLOTS, 128, 0, stream>>>(x, slot_token, expert_in);

    // GEMM1: h = relu(expert_in x wi[e])  [640x1024]x[1024x4096] per (e,g)
    wmma_gemm_kernel<true, true><<<dim3(F_DIM / BN, CAP / BM, E_EXP * G_GROUPS), 256, 0, stream>>>(
        expert_in, wi, hbuf, CAP, F_DIM, H_DIM,
        (size_t)CAP * H_DIM, (size_t)H_DIM * F_DIM, (size_t)CAP * F_DIM);

    // GEMM2: expert_out = h x wo[e]  [640x4096]x[4096x1024] per (e,g)
    wmma_gemm_kernel<false, false><<<dim3(H_DIM / BN, CAP / BM, E_EXP * G_GROUPS), 256, 0, stream>>>(
        hbuf, wo, expert_out, CAP, H_DIM, F_DIM,
        (size_t)CAP * F_DIM, (size_t)F_DIM * H_DIM, (size_t)CAP * H_DIM);

    combine_kernel<<<GT, 256, 0, stream>>>(token_assign, gates, expert_out, y);
}
